// SelfAttention_10574209483222
// MI455X (gfx1250) — matrix operand order repaired and checked by reading, not compiled, not run on hardware
//
#include <hip/hip_runtime.h>
#include <hip/hip_bf16.h>

// ---------- types ----------
typedef __attribute__((ext_vector_type(16))) __bf16          v16bf;
typedef __attribute__((ext_vector_type(16))) unsigned short  v16u;
typedef __attribute__((ext_vector_type(8)))  unsigned short  v8u;
typedef __attribute__((ext_vector_type(8)))  float           v8f;
typedef __attribute__((ext_vector_type(4)))  int             v4i;

#define AS_GLOBAL __attribute__((address_space(1)))
#define AS_LDS    __attribute__((address_space(3)))

union ABCast { v16u u; v16bf b; };

// ---------- CDNA5 async global->LDS (guarded; fallback keeps compile green) ----------
#if defined(__has_builtin)
#if __has_builtin(__builtin_amdgcn_global_load_async_to_lds_b128) && \
    __has_builtin(__builtin_amdgcn_s_wait_asynccnt)
#define HAVE_ASYNC_LDS 1
#endif
#endif

static __device__ __forceinline__ void stage16(unsigned short* dst, const unsigned short* src) {
#ifdef HAVE_ASYNC_LDS
    __builtin_amdgcn_global_load_async_to_lds_b128(
        (AS_GLOBAL v4i*)src,
        (AS_LDS    v4i*)dst,
        /*offset=*/0, /*cpol=*/0);
#else
    *(v8u*)dst = *(const v8u*)src;
#endif
}
static __device__ __forceinline__ void stage_fence() {
#ifdef HAVE_ASYNC_LDS
    __builtin_amdgcn_s_wait_asynccnt(0);
#endif
    __syncthreads();
}

// ---------- bf16 helpers (RNE) ----------
static __device__ __forceinline__ unsigned short f32_to_bf16(float f) {
    union { float f; unsigned u; } v; v.f = f;
    unsigned r = v.u + 0x7FFFu + ((v.u >> 16) & 1u);
    return (unsigned short)(r >> 16);
}

// ---------- WMMA wrapper: D = A(16x32 bf16) * B(32x16 bf16) + C(16x16 f32) ----------
static __device__ __forceinline__ v8f wmma_bf16(v16u a, v16u b, v8f c) {
    ABCast A; A.u = a;
    ABCast B; B.u = b;
    return __builtin_amdgcn_wmma_f32_16x16x32_bf16(
        false, A.b, false, B.b, (short)0, c, false, false);
}

// A fragment: lane owns row (lane&15); elems 0..7 = row[kBase+0..7], 8..15 = row[kBase+16..23]
static __device__ __forceinline__ v16u load_a_frag(const unsigned short* row, int kBase) {
    v8u lo = *(const v8u*)(row + kBase);
    v8u hi = *(const v8u*)(row + kBase + 16);
    return __builtin_shufflevector(lo, hi, 0,1,2,3,4,5,6,7,8,9,10,11,12,13,14,15);
}
// B fragment (Bt layout: row = output col n, contiguous along K): elems 0..7 = row[g*8+0..7], 8..15 = row[g*8+16..23]
static __device__ __forceinline__ v16u load_b_frag(const unsigned short* row, int g) {
    v8u lo = *(const v8u*)(row + g * 8);
    v8u hi = *(const v8u*)(row + g * 8 + 16);
    return __builtin_shufflevector(lo, hi, 0,1,2,3,4,5,6,7,8,9,10,11,12,13,14,15);
}

// ---------- kernel 0: fp32 -> bf16 ----------
__global__ void cvt_f32_bf16(const float* __restrict__ in, unsigned short* __restrict__ out, int n) {
    int i = blockIdx.x * blockDim.x + threadIdx.x;
    if (i < n) out[i] = f32_to_bf16(in[i]);
}

// =====================================================================================
// kernel 1: qkv = x @ w_in^T  — 8-wave cooperative 128x64 tile, LDS double-buffered B
// scatters q/k to [B,H,T,hd] and v transposed to [B,H,hd,T]
// =====================================================================================
__global__ __launch_bounds__(256)
void qkv_gemm(const unsigned short* __restrict__ xb,   // [8192][1024]
              const unsigned short* __restrict__ wib,  // [3072][1024]
              unsigned short* __restrict__ qv,
              unsigned short* __restrict__ kv,
              unsigned short* __restrict__ vtv) {
    __shared__ __align__(16) unsigned short Bs[2][64 * 32];

    const int tid  = threadIdx.x;
    const int lane = tid & 31;
    const int wv   = tid >> 5;          // wave 0..7
    const int col  = lane & 15;
    const int g    = lane >> 4;
    const int C    = 1024;
    const int m0   = blockIdx.y * 128 + wv * 16;   // token tile for this wave
    const int n0   = blockIdx.x * 64;              // output-feature macro tile

    const unsigned short* arow = xb + (m0 + col) * C;
    // B staging: 64 rows x 64B, 256 threads x 16B each
    const int srow = tid >> 2;                     // 0..63
    const int schk = (tid & 3) * 8;                // ushort offset in row
    const unsigned short* sgl = wib + (n0 + srow) * C + schk;
    unsigned short* sl0 = &Bs[0][srow * 32 + schk];
    unsigned short* sl1 = &Bs[1][srow * 32 + schk];

    stage16(sl0, sgl);                 // prologue: k-tile 0
    stage_fence();

    v8f acc[4] = {};
    v16u a = load_a_frag(arow, g * 8);
    for (int kt = 0; kt < 32; ++kt) {
        const int k0 = kt * 32;
        const int buf = kt & 1;
        if (kt + 1 < 32) {
            stage16(buf ? sl0 : sl1, sgl + k0 + 32);        // async next B tile
            __builtin_prefetch(arow + k0 + 64, 0, 0);
        }
        v16u aCur = a;
        if (kt + 1 < 32) a = load_a_frag(arow + k0 + 32, g * 8);  // pipeline next A
        #pragma unroll
        for (int sub = 0; sub < 4; ++sub) {
            v16u b = load_b_frag(&Bs[buf][(sub * 16 + col) * 32], g);
            acc[sub] = wmma_bf16(aCur, b, acc[sub]);
        }
        stage_fence();
    }

    #pragma unroll
    for (int sub = 0; sub < 4; ++sub) {
        int n = n0 + sub * 16 + col;
        int which = n >> 10;          // 0=q 1=k 2=v
        int c = n & 1023;
        int h = c >> 6, d = c & 63;
        #pragma unroll
        for (int r = 0; r < 8; ++r) {
            int tok = m0 + g * 8 + r;
            int b = tok >> 11, t = tok & 2047;
            int bh = b * 16 + h;
            unsigned short val = f32_to_bf16(acc[sub][r]);
            if (which == 0)      qv [(bh * 2048 + t) * 64 + d] = val;
            else if (which == 1) kv [(bh * 2048 + t) * 64 + d] = val;
            else                 vtv[(bh * 64 + d) * 2048 + t] = val;
        }
    }
}

// =====================================================================================
// kernel 2: causal flash attention, one wave per (b,h, 16-query tile)
// =====================================================================================
__global__ __launch_bounds__(32)
void attn(const unsigned short* __restrict__ qv,
          const unsigned short* __restrict__ kv,
          const unsigned short* __restrict__ vtv,
          unsigned short* __restrict__ yb) {
    __shared__ __align__(16) float          sS[16][32];
    __shared__ __align__(16) unsigned short pS[16][32];
    __shared__ float mRow[16], lRow[16], fRow[16];

    const int lane = threadIdx.x & 31;
    const int col  = lane & 15;
    const int g    = lane >> 4;
    const int bh   = blockIdx.x;       // 0..63
    const int qi   = blockIdx.y;       // 0..127
    const int q0   = qi * 16;
    const int T = 2048, HD = 64;
    const float scale = 0.125f;        // 1/sqrt(64)

    const unsigned short* qrow = qv + (bh * T + q0 + col) * HD;
    v16u aq0 = load_a_frag(qrow, g * 8);
    v16u aq1 = load_a_frag(qrow + 32, g * 8);

    if (lane < 16) { mRow[lane] = -1e30f; lRow[lane] = 0.0f; }
    __syncthreads();

    v8f accO[4] = {};
    const int nSteps = (q0 + 16 + 31) / 32;     // 32-key steps, causal bound
    for (int kt = 0; kt < nSteps; ++kt) {
        const int k0 = kt * 32;

        // ---- S = Q K^T (two 16x16 n-subtiles over 32 keys) ----
        v8f s0 = {}, s1 = {};
        {
            const unsigned short* krow0 = kv + (bh * T + k0 + col) * HD;
            const unsigned short* krow1 = krow0 + 16 * HD;
            v16u b00 = load_b_frag(krow0,      g);
            v16u b01 = load_b_frag(krow0 + 32, g);
            v16u b10 = load_b_frag(krow1,      g);
            v16u b11 = load_b_frag(krow1 + 32, g);
            s0 = wmma_bf16(aq0, b00, s0);
            s0 = wmma_bf16(aq1, b01, s0);
            s1 = wmma_bf16(aq0, b10, s1);
            s1 = wmma_bf16(aq1, b11, s1);
        }

        // ---- scale + causal mask -> LDS ----
        #pragma unroll
        for (int r = 0; r < 8; ++r) {
            int qr = q0 + g * 8 + r;
            float v0 = s0[r] * scale;
            float v1 = s1[r] * scale;
            if (k0 + col > qr)      v0 = -1e30f;
            if (k0 + 16 + col > qr) v1 = -1e30f;
            sS[g * 8 + r][col]      = v0;
            sS[g * 8 + r][16 + col] = v1;
        }
        __syncthreads();

        // ---- online softmax: one lane per query row ----
        if (lane < 16) {
            float m = mRow[lane], l = lRow[lane];
            float rmax = m;
            #pragma unroll
            for (int j = 0; j < 32; ++j) rmax = fmaxf(rmax, sS[lane][j]);
            float f = __expf(m - rmax);
            float sum = 0.0f;
            #pragma unroll
            for (int j = 0; j < 32; ++j) {
                float p = __expf(sS[lane][j] - rmax);
                pS[lane][j] = f32_to_bf16(p);
                sum += p;
            }
            mRow[lane] = rmax;
            lRow[lane] = l * f + sum;
            fRow[lane] = f;
        }
        __syncthreads();

        // ---- rescale O accumulators ----
        #pragma unroll
        for (int r = 0; r < 8; ++r) {
            float f = fRow[g * 8 + r];
            #pragma unroll
            for (int d = 0; d < 4; ++d) accO[d][r] *= f;
        }

        // ---- O += P @ V (P in A layout from LDS; V rows contiguous in vT) ----
        v16u ap = load_a_frag(&pS[col][0], g * 8);
        #pragma unroll
        for (int d = 0; d < 4; ++d) {
            const unsigned short* vrow = vtv + (bh * HD + d * 16 + col) * T + k0;
            accO[d] = wmma_bf16(ap, load_b_frag(vrow, g), accO[d]);
        }
        __syncthreads();
    }

    // ---- normalize, write y [B,T,C] bf16 ----
    const int b = bh >> 4, h = bh & 15;
    #pragma unroll
    for (int d = 0; d < 4; ++d) {
        #pragma unroll
        for (int r = 0; r < 8; ++r) {
            int t = q0 + g * 8 + r;
            float val = accO[d][r] / lRow[g * 8 + r];
            yb[(b * T + t) * 1024 + h * 64 + d * 16 + col] = f32_to_bf16(val);
        }
    }
}

// =====================================================================================
// kernel 3: out = y @ w_out^T (fp32 result) — same cooperative scheme as kernel 1
// =====================================================================================
__global__ __launch_bounds__(256)
void out_gemm(const unsigned short* __restrict__ yb,   // [8192][1024]
              const unsigned short* __restrict__ wob,  // [1024][1024]
              float* __restrict__ out) {
    __shared__ __align__(16) unsigned short Bs[2][64 * 32];

    const int tid  = threadIdx.x;
    const int lane = tid & 31;
    const int wv   = tid >> 5;
    const int col  = lane & 15;
    const int g    = lane >> 4;
    const int C    = 1024;
    const int m0   = blockIdx.y * 128 + wv * 16;
    const int n0   = blockIdx.x * 64;

    const unsigned short* arow = yb + (m0 + col) * C;
    const int srow = tid >> 2;
    const int schk = (tid & 3) * 8;
    const unsigned short* sgl = wob + (n0 + srow) * C + schk;
    unsigned short* sl0 = &Bs[0][srow * 32 + schk];
    unsigned short* sl1 = &Bs[1][srow * 32 + schk];

    stage16(sl0, sgl);
    stage_fence();

    v8f acc[4] = {};
    v16u a = load_a_frag(arow, g * 8);
    for (int kt = 0; kt < 32; ++kt) {
        const int k0 = kt * 32;
        const int buf = kt & 1;
        if (kt + 1 < 32) {
            stage16(buf ? sl0 : sl1, sgl + k0 + 32);
            __builtin_prefetch(arow + k0 + 64, 0, 0);
        }
        v16u aCur = a;
        if (kt + 1 < 32) a = load_a_frag(arow + k0 + 32, g * 8);
        #pragma unroll
        for (int sub = 0; sub < 4; ++sub) {
            v16u b = load_b_frag(&Bs[buf][(sub * 16 + col) * 32], g);
            acc[sub] = wmma_bf16(aCur, b, acc[sub]);
        }
        stage_fence();
    }

    #pragma unroll
    for (int sub = 0; sub < 4; ++sub) {
        int n = n0 + sub * 16 + col;
        #pragma unroll
        for (int r = 0; r < 8; ++r)
            out[(m0 + g * 8 + r) * C + n] = acc[sub][r];
    }
}

// ---------- launch ----------
extern "C" void kernel_launch(void* const* d_in, const int* in_sizes, int n_in,
                              void* d_out, int out_size, void* d_ws, size_t ws_size,
                              hipStream_t stream) {
    (void)in_sizes; (void)n_in; (void)out_size; (void)ws_size;
    const float* x     = (const float*)d_in[0];
    const float* w_in  = (const float*)d_in[1];
    const float* w_out = (const float*)d_in[2];
    float* out = (float*)d_out;

    const int B = 4, T = 2048, C = 1024, H = 16;
    const size_t NT = (size_t)B * T;        // 8192 tokens

    char* ws = (char*)d_ws;
    size_t off = 0;
    auto alloc = [&](size_t bytes) {
        char* p = ws + off;
        off += (bytes + 255) & ~(size_t)255;
        return p;
    };
    unsigned short* xb  = (unsigned short*)alloc(NT * C * 2);
    unsigned short* wib = (unsigned short*)alloc((size_t)3 * C * C * 2);
    unsigned short* wob = (unsigned short*)alloc((size_t)C * C * 2);
    unsigned short* qb  = (unsigned short*)alloc(NT * C * 2);
    unsigned short* kb  = (unsigned short*)alloc(NT * C * 2);
    unsigned short* vtb = (unsigned short*)alloc(NT * C * 2);
    unsigned short* ybf = (unsigned short*)alloc(NT * C * 2);

    int n1 = (int)(NT * C);
    int n2 = 3 * C * C;
    int n3 = C * C;
    cvt_f32_bf16<<<(n1 + 255) / 256, 256, 0, stream>>>(x, xb, n1);
    cvt_f32_bf16<<<(n2 + 255) / 256, 256, 0, stream>>>(w_in, wib, n2);
    cvt_f32_bf16<<<(n3 + 255) / 256, 256, 0, stream>>>(w_out, wob, n3);

    dim3 g1(3 * C / 64, NT / 128);          // 48 x 64, 256 threads
    qkv_gemm<<<g1, 256, 0, stream>>>(xb, wib, qb, kb, vtb);

    dim3 g2(B * H, T / 16);                 // 64 x 128, 32 threads
    attn<<<g2, 32, 0, stream>>>(qb, kb, vtb, ybf);

    dim3 g3(C / 64, NT / 128);              // 16 x 64, 256 threads
    out_gemm<<<g3, 256, 0, stream>>>(ybf, wob, out);
}
